// BasicLayer_49478023250473
// MI455X (gfx1250) — matrix operand order repaired and checked by reading, not compiled, not run on hardware
//
#include <hip/hip_runtime.h>
#include <hip/hip_bf16.h>

// ---------------- problem constants (match reference) ----------------
#define BSZ     2
#define LSEQ    2048
#define DMODEL  1024
#define DINNER  2048
#define DSTATE  16
#define DTRANK  64
#define DCONV   4
#define DEPTH   2
#define MROWS   (BSZ * LSEQ)          // 4096 token rows

typedef __bf16 bf16;
typedef __attribute__((ext_vector_type(16))) __bf16 v16bf;
typedef __attribute__((ext_vector_type(8)))  float  v8f;

// ---------------------------------------------------------------------
// Generic f32 -> bf16 conversion (weights, activations)
// ---------------------------------------------------------------------
__global__ void convert_f32_bf16(const float* __restrict__ src,
                                 bf16* __restrict__ dst, int n) {
  int i = blockIdx.x * blockDim.x + threadIdx.x;
  if (i < n) dst[i] = (bf16)src[i];
}

// ---------------------------------------------------------------------
// LayerNorm over DMODEL, emit bf16 normalized rows for the WMMA GEMM.
// One block (256 threads) per token row.
// ---------------------------------------------------------------------
__global__ __launch_bounds__(256) void layernorm_bf16(
    const float* __restrict__ X, const float* __restrict__ w,
    const float* __restrict__ b, bf16* __restrict__ Y) {
  __shared__ float red[256];
  const int row = blockIdx.x;
  const float* xr = X + (size_t)row * DMODEL;

  float s = 0.f;
  for (int c = threadIdx.x; c < DMODEL; c += 256) s += xr[c];
  red[threadIdx.x] = s; __syncthreads();
  for (int off = 128; off > 0; off >>= 1) {
    if (threadIdx.x < off) red[threadIdx.x] += red[threadIdx.x + off];
    __syncthreads();
  }
  const float mu = red[0] * (1.0f / DMODEL);
  __syncthreads();

  float v = 0.f;
  for (int c = threadIdx.x; c < DMODEL; c += 256) {
    float d = xr[c] - mu; v += d * d;
  }
  red[threadIdx.x] = v; __syncthreads();
  for (int off = 128; off > 0; off >>= 1) {
    if (threadIdx.x < off) red[threadIdx.x] += red[threadIdx.x + off];
    __syncthreads();
  }
  const float rstd = rsqrtf(red[0] * (1.0f / DMODEL) + 1e-5f);

  bf16* yr = Y + (size_t)row * DMODEL;
  for (int c = threadIdx.x; c < DMODEL; c += 256)
    yr[c] = (bf16)(((xr[c] - mu) * rstd) * w[c] + b[c]);
}

// ---------------------------------------------------------------------
// Fragment loaders (layouts per CDNA5 ISA 7.12.2, wave32):
//   A 16x32 bf16: lane = hi*16 + m; the lane's 16 elements are the two
//     contiguous 8-element chunks at K offsets (8*hi) and (16 + 8*hi).
//   B 32x16 bf16: lane = hi*16 + n; the lane's 16 elements are the two
//     contiguous 8-element chunks W[n][k0 + 8*hi ..] and W[n][k0 + 16 + 8*hi ..].
// ---------------------------------------------------------------------
__device__ __forceinline__ v16bf load_fragA(const bf16* p /* = Arow+8*hi+k0 */) {
  union { v16bf v; uint4 q[2]; } f;
  f.q[0] = *reinterpret_cast<const uint4*>(p);
  f.q[1] = *reinterpret_cast<const uint4*>(p + 16);
  return f.v;
}
__device__ __forceinline__ v16bf load_fragB(const bf16* p /* = Wrow+8*hi+k0 */) {
  union { v16bf v; uint4 q[2]; } f;
  f.q[0] = *reinterpret_cast<const uint4*>(p);
  f.q[1] = *reinterpret_cast<const uint4*>(p + 16);
  return f.v;
}

// ---------------------------------------------------------------------
// Register-blocked WMMA bf16 GEMM:  C[M,N] = A[M,K] * W[N,K]^T (f32 acc)
//   Each wave computes an (MT x NT) grid of 16x16 tiles: A fragments are
//   reused NT times, B fragments MT times -> MT+NT b128-pairs feed MT*NT
//   WMMAs per k-step (vs 4 loads per WMMA unblocked).
//   8 waves per 256-thread block tile along M. Exact grids -> EXEC all 1s.
// ---------------------------------------------------------------------
template <int MT, int NT>
__global__ __launch_bounds__(256) void gemm_bf16_wmma(
    const bf16* __restrict__ A, const bf16* __restrict__ W,
    float* __restrict__ C, int M, int N, int K) {
  const int lane = threadIdx.x & 31;
  const int wave = threadIdx.x >> 5;
  const int n0 = blockIdx.x * (16 * NT);
  const int m0 = (blockIdx.y * 8 + wave) * (16 * MT);
  const int hi  = lane >> 4;
  const int low = lane & 15;

  const bf16* Arow[MT];
#pragma unroll
  for (int i = 0; i < MT; ++i)
    Arow[i] = A + (size_t)(m0 + 16 * i + low) * K + 8 * hi;
  const bf16* Wrow[NT];
#pragma unroll
  for (int j = 0; j < NT; ++j)
    Wrow[j] = W + (size_t)(n0 + 16 * j + low) * K + 8 * hi;

  v8f acc[MT][NT];
#pragma unroll
  for (int i = 0; i < MT; ++i)
#pragma unroll
    for (int j = 0; j < NT; ++j)
#pragma unroll
      for (int r = 0; r < 8; ++r) acc[i][j][r] = 0.f;

  for (int k0 = 0; k0 < K; k0 += 32) {
    v16bf af[MT];
#pragma unroll
    for (int i = 0; i < MT; ++i) af[i] = load_fragA(Arow[i] + k0);
    v16bf wf[NT];
#pragma unroll
    for (int j = 0; j < NT; ++j) wf[j] = load_fragB(Wrow[j] + k0);
#pragma unroll
    for (int i = 0; i < MT; ++i)
#pragma unroll
      for (int j = 0; j < NT; ++j)
        acc[i][j] = __builtin_amdgcn_wmma_f32_16x16x32_bf16(
            /*neg_a=*/false, af[i], /*neg_b=*/false, wf[j],
            /*c_mod=*/(short)0, acc[i][j],
            /*reuse_a=*/false, /*reuse_b=*/false);
  }

#pragma unroll
  for (int i = 0; i < MT; ++i)
#pragma unroll
    for (int j = 0; j < NT; ++j) {
      float* Crow = C + (size_t)(m0 + 16 * i) * N + (n0 + 16 * j + low);
#pragma unroll
      for (int r = 0; r < 8; ++r)
        Crow[(size_t)(r + 8 * hi) * N] = acc[i][j][r];
    }
}

// ---------------------------------------------------------------------
// Causal depthwise conv1d (k=4) + bias + SiLU on u = xz[:, :DINNER].
// Emits both f32 (for the scan) and bf16 (for the x_proj GEMM).
// ---------------------------------------------------------------------
__global__ void conv_silu(const float* __restrict__ xz,
                          const float* __restrict__ cw,
                          const float* __restrict__ cb,
                          float* __restrict__ u, bf16* __restrict__ ub) {
  const int idx = blockIdx.x * blockDim.x + threadIdx.x;
  if (idx >= MROWS * DINNER) return;
  const int d   = idx % DINNER;
  const int row = idx / DINNER;          // b*L + l
  const int l   = row % LSEQ;

  float acc = cb[d];
#pragma unroll
  for (int t = 0; t < DCONV; ++t) {
    const int ll = l - (DCONV - 1) + t;
    if (ll >= 0)
      acc += cw[d * DCONV + t] *
             xz[(size_t)(row - (DCONV - 1) + t) * (2 * DINNER) + d];
  }
  const float s = acc / (1.f + __expf(-acc));   // silu
  u[idx]  = s;
  ub[idx] = (bf16)s;
}

// ---------------------------------------------------------------------
// Split x_dbl (MROWS x 96) -> dt_r (bf16, 64), Bm (f32, 16), Cm (f32, 16)
// ---------------------------------------------------------------------
__global__ void split_xdbl(const float* __restrict__ xd,
                           bf16* __restrict__ dtr,
                           float* __restrict__ Bm, float* __restrict__ Cm) {
  const int idx = blockIdx.x * blockDim.x + threadIdx.x;
  if (idx >= MROWS * (DTRANK + 2 * DSTATE)) return;
  const int row = idx / (DTRANK + 2 * DSTATE);
  const int c   = idx % (DTRANK + 2 * DSTATE);
  const float v = xd[idx];
  if (c < DTRANK)               dtr[row * DTRANK + c] = (bf16)v;
  else if (c < DTRANK + DSTATE) Bm[row * DSTATE + (c - DTRANK)] = v;
  else                          Cm[row * DSTATE + (c - DTRANK - DSTATE)] = v;
}

// ---------------------------------------------------------------------
// dt = softplus(dt_pre + bias)   (in place)
// ---------------------------------------------------------------------
__global__ void softplus_bias(float* __restrict__ dt,
                              const float* __restrict__ b) {
  const int idx = blockIdx.x * blockDim.x + threadIdx.x;
  if (idx >= MROWS * DINNER) return;
  const float v = dt[idx] + b[idx % DINNER];
  dt[idx] = (v > 20.f) ? v : __logf(1.f + __expf(v));
}

// ---------------------------------------------------------------------
// Selective scan: one thread per (batch, channel), 16-state recurrence,
// sequential over L. y written in place over u (read-then-write).
// ---------------------------------------------------------------------
__global__ void scan_kernel(const float* __restrict__ dt,
                            const float* __restrict__ Bm,
                            const float* __restrict__ Cm,
                            const float* __restrict__ Alog,
                            const float* __restrict__ Dp,
                            float* __restrict__ uy /* u in, y out */) {
  const int t = blockIdx.x * blockDim.x + threadIdx.x;
  if (t >= BSZ * DINNER) return;
  const int b = t / DINNER;
  const int d = t % DINNER;

  float A[DSTATE], h[DSTATE];
#pragma unroll
  for (int n = 0; n < DSTATE; ++n) {
    A[n] = -__expf(Alog[d * DSTATE + n]);
    h[n] = 0.f;
  }
  const float Dv = Dp[d];

  for (int l = 0; l < LSEQ; ++l) {
    const size_t r  = (size_t)(b * LSEQ + l);
    const float uv  = uy[r * DINNER + d];
    const float dv  = dt[r * DINNER + d];
    const float dvu = dv * uv;
    const float* Bt = Bm + r * DSTATE;
    const float* Ct = Cm + r * DSTATE;
    float acc = 0.f;
#pragma unroll
    for (int n = 0; n < DSTATE; ++n) {
      h[n] = __expf(dv * A[n]) * h[n] + dvu * Bt[n];
      acc += h[n] * Ct[n];
    }
    uy[r * DINNER + d] = acc + uv * Dv;
  }
}

// ---------------------------------------------------------------------
// y * silu(z) -> bf16 for out_proj GEMM
// ---------------------------------------------------------------------
__global__ void gate_bf16(const float* __restrict__ y,
                          const float* __restrict__ xz,
                          bf16* __restrict__ yb) {
  const int idx = blockIdx.x * blockDim.x + threadIdx.x;
  if (idx >= MROWS * DINNER) return;
  const int row = idx / DINNER;
  const int d   = idx % DINNER;
  const float z = xz[(size_t)row * (2 * DINNER) + DINNER + d];
  const float g = z / (1.f + __expf(-z));
  yb[idx] = (bf16)(y[idx] * g);
}

// ---------------------------------------------------------------------
static inline size_t align256(size_t x) { return (x + 255) & ~(size_t)255; }

extern "C" void kernel_launch(void* const* d_in, const int* in_sizes, int n_in,
                              void* d_out, int out_size, void* d_ws, size_t ws_size,
                              hipStream_t stream) {
  (void)in_sizes; (void)n_in; (void)out_size; (void)ws_size;

  const float* x         = (const float*)d_in[0];
  // d_in[1] = x_size (unused in the math)
  const float* norm_w    = (const float*)d_in[2];
  const float* norm_b    = (const float*)d_in[3];
  const float* in_proj_w = (const float*)d_in[4];
  const float* conv_w    = (const float*)d_in[5];
  const float* conv_b    = (const float*)d_in[6];
  const float* x_proj_w  = (const float*)d_in[7];
  const float* dt_proj_w = (const float*)d_in[8];
  const float* dt_proj_b = (const float*)d_in[9];
  const float* A_log     = (const float*)d_in[10];
  const float* Dw        = (const float*)d_in[11];
  const float* out_proj_w= (const float*)d_in[12];

  // ---- bump-allocate workspace ----
  char* ws = (char*)d_ws;
  size_t off = 0;
  auto alloc = [&](size_t bytes) -> char* {
    char* p = ws + off; off = align256(off + bytes); return p;
  };
  const int NDBL = DTRANK + 2 * DSTATE;  // 96

  bf16* wb_in  = (bf16*)alloc((size_t)DEPTH * 2 * DINNER * DMODEL * sizeof(bf16));
  bf16* wb_xp  = (bf16*)alloc((size_t)DEPTH * NDBL * DINNER * sizeof(bf16));
  bf16* wb_dt  = (bf16*)alloc((size_t)DEPTH * DINNER * DTRANK * sizeof(bf16));
  bf16* wb_out = (bf16*)alloc((size_t)DEPTH * DMODEL * DINNER * sizeof(bf16));
  bf16* hn_b   = (bf16*)alloc((size_t)MROWS * DMODEL * sizeof(bf16));
  float* xz    = (float*)alloc((size_t)MROWS * 2 * DINNER * sizeof(float));
  float* u     = (float*)alloc((size_t)MROWS * DINNER * sizeof(float)); // also y
  bf16*  ub    = (bf16*) alloc((size_t)MROWS * DINNER * sizeof(bf16));  // also yb
  float* xdbl  = (float*)alloc((size_t)MROWS * NDBL * sizeof(float));
  bf16*  dtr   = (bf16*) alloc((size_t)MROWS * DTRANK * sizeof(bf16));
  float* Bm    = (float*)alloc((size_t)MROWS * DSTATE * sizeof(float));
  float* Cm    = (float*)alloc((size_t)MROWS * DSTATE * sizeof(float));
  float* dtf   = (float*)alloc((size_t)MROWS * DINNER * sizeof(float));
  float* hbuf  = (float*)alloc((size_t)MROWS * DMODEL * sizeof(float));

  const int TPB = 256;
  auto blocks = [](long n, int t) { return (unsigned)((n + t - 1) / t); };

  // ---- convert all weights to bf16 once ----
  {
    long n;
    n = (long)DEPTH * 2 * DINNER * DMODEL;
    convert_f32_bf16<<<blocks(n, TPB), TPB, 0, stream>>>(in_proj_w, wb_in, (int)n);
    n = (long)DEPTH * NDBL * DINNER;
    convert_f32_bf16<<<blocks(n, TPB), TPB, 0, stream>>>(x_proj_w, wb_xp, (int)n);
    n = (long)DEPTH * DINNER * DTRANK;
    convert_f32_bf16<<<blocks(n, TPB), TPB, 0, stream>>>(dt_proj_w, wb_dt, (int)n);
    n = (long)DEPTH * DMODEL * DINNER;
    convert_f32_bf16<<<blocks(n, TPB), TPB, 0, stream>>>(out_proj_w, wb_out, (int)n);
  }

  // Block covers 8 waves * MT(=2) tiles = 256 rows of M.
  const int MT = 2;
  const dim3 gridM(1, MROWS / (16 * MT * 8));  // y-dim shared by all GEMMs

  for (int i = 0; i < DEPTH; ++i) {
    const float* hin  = (i == 0) ? x : hbuf;
    float*       hout = (i == DEPTH - 1) ? (float*)d_out : hbuf;

    // 1) LayerNorm -> bf16
    layernorm_bf16<<<MROWS, TPB, 0, stream>>>(
        hin, norm_w + (size_t)i * DMODEL, norm_b + (size_t)i * DMODEL, hn_b);

    // 2) in_proj: xz = hn * Win^T   (4096 x 4096 x 1024), 32x64 per wave
    gemm_bf16_wmma<2, 4><<<dim3(2 * DINNER / 64, gridM.y), TPB, 0, stream>>>(
        hn_b, wb_in + (size_t)i * 2 * DINNER * DMODEL, xz,
        MROWS, 2 * DINNER, DMODEL);

    // 3) causal depthwise conv + SiLU
    conv_silu<<<blocks((long)MROWS * DINNER, TPB), TPB, 0, stream>>>(
        xz, conv_w + (size_t)i * DINNER * DCONV, conv_b + (size_t)i * DINNER,
        u, ub);

    // 4) x_proj: x_dbl = u * Wxp^T  (4096 x 96 x 2048), 32x32 per wave
    gemm_bf16_wmma<2, 2><<<dim3(NDBL / 32, gridM.y), TPB, 0, stream>>>(
        ub, wb_xp + (size_t)i * NDBL * DINNER, xdbl, MROWS, NDBL, DINNER);

    // 5) split into dt_r (bf16), Bm, Cm
    split_xdbl<<<blocks((long)MROWS * NDBL, TPB), TPB, 0, stream>>>(
        xdbl, dtr, Bm, Cm);

    // 6) dt_proj: dt_pre = dt_r * Wdt^T (4096 x 2048 x 64)
    gemm_bf16_wmma<2, 4><<<dim3(DINNER / 64, gridM.y), TPB, 0, stream>>>(
        dtr, wb_dt + (size_t)i * DINNER * DTRANK, dtf, MROWS, DINNER, DTRANK);

    // 7) dt = softplus(dt_pre + b)
    softplus_bias<<<blocks((long)MROWS * DINNER, TPB), TPB, 0, stream>>>(
        dtf, dt_proj_b + (size_t)i * DINNER);

    // 8) selective scan (y overwrites u in place)
    scan_kernel<<<blocks((long)BSZ * DINNER, TPB), TPB, 0, stream>>>(
        dtf, Bm, Cm, A_log + (size_t)i * DINNER * DSTATE,
        Dw + (size_t)i * DINNER, u);

    // 9) gate: yb = bf16(y * silu(z))
    gate_bf16<<<blocks((long)MROWS * DINNER, TPB), TPB, 0, stream>>>(u, xz, ub);

    // 10) out_proj: h = yb * Wout^T (4096 x 1024 x 2048)
    gemm_bf16_wmma<2, 4><<<dim3(DMODEL / 64, gridM.y), TPB, 0, stream>>>(
        ub, wb_out + (size_t)i * DMODEL * DINNER, hout, MROWS, DMODEL, DINNER);
  }
}
